// MHMoE_37177236914789
// MI455X (gfx1250) — matrix operand order repaired and checked by reading, not compiled, not run on hardware
//
#include <hip/hip_runtime.h>

// ---------------- types ----------------
typedef __attribute__((ext_vector_type(16))) __bf16        bf16x16;
typedef __attribute__((ext_vector_type(8)))  float         f32x8;
typedef __attribute__((ext_vector_type(8)))  unsigned int  u32x8;
typedef __attribute__((ext_vector_type(4)))  int           v4i;

#define BS_    4096
#define H_     2048
#define NH_    4
#define HD_    512
#define NEXP_  8
#define INTER_ 2048
#define TOKENS_ 16384   // BS*NH

#if __has_builtin(__builtin_amdgcn_global_load_async_to_lds_b128)
#define ASYNC_LDS_OK 1
#else
#define ASYNC_LDS_OK 0
#endif

// ---------------- helpers ----------------
__device__ __forceinline__ unsigned short f2bf(float f) {
    unsigned int u = __float_as_uint(f);
    unsigned int r = u + 0x7FFFu + ((u >> 16) & 1u);   // round-to-nearest-even
    return (unsigned short)(r >> 16);
}
__device__ __forceinline__ float bf2f(unsigned short h) {
    return __uint_as_float(((unsigned int)h) << 16);
}

// Raw fragment loads (kept as u32x8 so pipelined copies are plain moves).
// A-operand (16x32 bf16, MxK): lane L = row m=L&15; K = k0+16*(L>>4) .. +15 contiguous.
__device__ __forceinline__ u32x8 load_a_raw(const unsigned short* row, int k0, int lh) {
    int r0 = k0 + lh * 16;
    uint4 lo = *(const uint4*)(row + r0);
    uint4 hi = *(const uint4*)(row + r0 + 8);
    u32x8 u;
    u[0] = lo.x; u[1] = lo.y; u[2] = lo.z; u[3] = lo.w;
    u[4] = hi.x; u[5] = hi.y; u[6] = hi.z; u[7] = hi.w;
    return u;
}
// B-operand (32x16 bf16, KxN), B[k][n] = W[n][k] with W row-major [N,K]:
// lane L = column n=L&15; K = k0+16*(L>>4) .. +15 contiguous in W's row.
__device__ __forceinline__ u32x8 load_b_raw(const unsigned short* wrow, int k0, int lh) {
    int kb = k0 + lh * 16;
    uint4 lo = *(const uint4*)(wrow + kb);
    uint4 hi = *(const uint4*)(wrow + kb + 8);
    u32x8 u;
    u[0] = lo.x; u[1] = lo.y; u[2] = lo.z; u[3] = lo.w;
    u[4] = hi.x; u[5] = hi.y; u[6] = hi.z; u[7] = hi.w;
    return u;
}
__device__ __forceinline__ bf16x16 as_bf(u32x8 u) { return __builtin_bit_cast(bf16x16, u); }
__device__ __forceinline__ f32x8 wmma_bf16(u32x8 a, u32x8 b, f32x8 c) {
    return __builtin_amdgcn_wmma_f32_16x16x32_bf16(false, as_bf(a), false, as_bf(b),
                                                   (short)0, c, false, false);
}

// ---------------- conversion kernels ----------------
__global__ void k_convert_bf16(const float* __restrict__ src,
                               unsigned short* __restrict__ dst, long n) {
    for (long o = (long)blockIdx.x * blockDim.x + threadIdx.x; o < n;
         o += (long)gridDim.x * blockDim.x)
        dst[o] = f2bf(src[o]);
}
// src: [B, rows, cols] f32  ->  dst: [B, cols, rows] bf16
__global__ void k_convert_transpose_bf16(const float* __restrict__ src,
                                         unsigned short* __restrict__ dst,
                                         int rows, int cols, long n) {
    for (long o = (long)blockIdx.x * blockDim.x + threadIdx.x; o < n;
         o += (long)gridDim.x * blockDim.x) {
        int  r = (int)(o % rows);
        long t = o / rows;
        int  c = (int)(t % cols);
        int  b = (int)(t / cols);
        dst[o] = f2bf(src[((long)b * rows + r) * cols + c]);
    }
}
__global__ void k_zero_cnt(int* cnt) {
    if (threadIdx.x < NEXP_) cnt[threadIdx.x] = 0;
}

// ---------------- generic GEMM: C[M,N] = A[M,K] @ B[N,K]^T + bias ----------------
// block = 256 threads = 8 waves; tile M=64 (4 wave-rows x 16), N=128 (2 wave-cols x 64)
// Software-pipelined: next k-step's A/B fragments are loaded before this step's WMMAs.
template <int OUT_BF16>
__global__ __launch_bounds__(256) void k_gemm_nt(const unsigned short* __restrict__ A,
                                                 const unsigned short* __restrict__ B,
                                                 const float* __restrict__ bias,
                                                 void* __restrict__ C,
                                                 int M, int N, int K) {
    const int wave = threadIdx.x >> 5, lane = threadIdx.x & 31;
    const int lm = lane & 15, lh = lane >> 4;
    const int wm = wave & 3, wn = wave >> 2;
    const int m0 = blockIdx.x * 64 + wm * 16;
    const int n0 = blockIdx.y * 128 + wn * 64;

    const unsigned short* arow = A + (size_t)(m0 + lm) * K;
    const unsigned short* brow[4];
    for (int nt = 0; nt < 4; ++nt) brow[nt] = B + (size_t)(n0 + nt * 16 + lm) * K;

    f32x8 acc[4] = {};
    u32x8 a_cur = load_a_raw(arow, 0, lh);
    u32x8 b_cur[4];
    for (int nt = 0; nt < 4; ++nt) b_cur[nt] = load_b_raw(brow[nt], 0, lh);

    for (int k0 = 0; k0 < K - 32; k0 += 32) {
        const int kn = k0 + 32;
        if ((k0 & 127) == 0 && k0 + 256 < K) {        // L2 prefetch a bit ahead
            __builtin_prefetch(arow + k0 + 256, 0, 3);
            __builtin_prefetch(brow[0] + k0 + 256, 0, 3);
            __builtin_prefetch(brow[1] + k0 + 256, 0, 3);
            __builtin_prefetch(brow[2] + k0 + 256, 0, 3);
            __builtin_prefetch(brow[3] + k0 + 256, 0, 3);
        }
        u32x8 a_nxt = load_a_raw(arow, kn, lh);
        u32x8 b_nxt[4];
        for (int nt = 0; nt < 4; ++nt) b_nxt[nt] = load_b_raw(brow[nt], kn, lh);
        for (int nt = 0; nt < 4; ++nt) acc[nt] = wmma_bf16(a_cur, b_cur[nt], acc[nt]);
        a_cur = a_nxt;
        for (int nt = 0; nt < 4; ++nt) b_cur[nt] = b_nxt[nt];
    }
    for (int nt = 0; nt < 4; ++nt) acc[nt] = wmma_bf16(a_cur, b_cur[nt], acc[nt]);

    for (int nt = 0; nt < 4; ++nt) {
        int n = n0 + nt * 16 + lm;
        float bv = bias ? bias[n] : 0.0f;
        for (int v = 0; v < 8; ++v) {
            int m = m0 + v + 8 * lh;
            float val = acc[nt][v] + bv;
            if (OUT_BF16)
                ((unsigned short*)C)[(size_t)m * N + n] = f2bf(val);
            else
                ((float*)C)[(size_t)m * N + n] = val;
        }
    }
}

// ---------------- router: softmax over experts, top-2, compaction ----------------
__global__ __launch_bounds__(128) void k_router(const unsigned short* __restrict__ h_bf,
                                                const float* __restrict__ emb,
                                                int* __restrict__ cnt,
                                                int* __restrict__ tok_list,
                                                float* __restrict__ w_list,
                                                int* __restrict__ slot_list) {
    const int wave = threadIdx.x >> 5, lane = threadIdx.x & 31;
    const int tok = blockIdx.x * 4 + wave;
    if (tok >= TOKENS_) return;
    const unsigned short* hrow = h_bf + (size_t)tok * HD_;

    float logits[NEXP_];
    for (int e = 0; e < NEXP_; ++e) {
        const float* erow = emb + e * HD_;
        float s = 0.0f;
        for (int k = lane; k < HD_; k += 32) s += bf2f(hrow[k]) * erow[k];
        for (int off = 16; off > 0; off >>= 1) s += __shfl_xor(s, off, 32);
        logits[e] = s;
    }
    if (lane == 0) {
        float mx = logits[0];
        for (int e = 1; e < NEXP_; ++e) mx = fmaxf(mx, logits[e]);
        float p[NEXP_], sum = 0.0f;
        for (int e = 0; e < NEXP_; ++e) { p[e] = __expf(logits[e] - mx); sum += p[e]; }
        float inv = 1.0f / sum;
        for (int e = 0; e < NEXP_; ++e) p[e] *= inv;
        int e0 = 0;
        for (int e = 1; e < NEXP_; ++e) if (p[e] > p[e0]) e0 = e;
        int e1 = -1;
        for (int e = 0; e < NEXP_; ++e)
            if (e != e0 && (e1 < 0 || p[e] > p[e1])) e1 = e;
        int p0 = atomicAdd(&cnt[e0], 1);
        tok_list[e0 * TOKENS_ + p0] = tok;
        w_list[e0 * TOKENS_ + p0] = p[e0];
        slot_list[e0 * TOKENS_ + p0] = 0;
        int p1 = atomicAdd(&cnt[e1], 1);
        tok_list[e1 * TOKENS_ + p1] = tok;
        w_list[e1 * TOKENS_ + p1] = p[e1];
        slot_list[e1 * TOKENS_ + p1] = 1;
    }
}

// ---------------- fused expert FFN over compacted token tiles ----------------
// block = 256 threads = 8 waves; 16 gathered tokens staged into LDS (async path);
// each wave owns 64 of the 512 output columns; INTER in 128-wide chunks (ReLU^2 in LDS).
#define ACT_LDW 136  // 128 + 8 pad
#define SH_LDW  520  // 512 + 8 pad (1040 B row stride: 16B-aligned, bank-staggered)
__global__ __launch_bounds__(256) void k_ffn_expert(const unsigned short* __restrict__ h_bf,
                                                    const unsigned short* __restrict__ upT,
                                                    const unsigned short* __restrict__ dnT,
                                                    const int* __restrict__ cnt,
                                                    const int* __restrict__ tok_list,
                                                    const float* __restrict__ w_list,
                                                    const int* __restrict__ slot_list,
                                                    float* __restrict__ slots) {
    const int e = blockIdx.y;
    const int t0 = blockIdx.x * 16;
    const int count = cnt[e];
    if (t0 >= count) return;

    __shared__ int            s_tok[16];
    __shared__ float          s_w[16];
    __shared__ int            s_slot[16];
    __shared__ unsigned short s_h[16 * SH_LDW];
    __shared__ unsigned short s_act[16 * ACT_LDW];

    const int tid = threadIdx.x;
    if (tid < 16) {
        int idx = t0 + tid;
        bool valid = idx < count;
        s_tok[tid]  = valid ? tok_list[e * TOKENS_ + idx] : 0;
        s_w[tid]    = valid ? w_list[e * TOKENS_ + idx] : 0.0f;
        s_slot[tid] = valid ? slot_list[e * TOKENS_ + idx] : -1;
    }
    __syncthreads();

    // ---- stage the 16 gathered h rows into LDS (16 KB), async global->LDS if available
    for (int c = tid; c < 1024; c += 256) {           // 1024 x 16B chunks
        int m = c >> 6, j = (c & 63) * 8;
        const unsigned short* g = h_bf + (size_t)s_tok[m] * HD_ + j;
        unsigned short* l = &s_h[m * SH_LDW + j];
#if ASYNC_LDS_OK
        __builtin_amdgcn_global_load_async_to_lds_b128(
            (__attribute__((address_space(1))) v4i*)g,
            (__attribute__((address_space(3))) v4i*)l, 0, 0);
#else
        *(uint4*)l = *(const uint4*)g;
#endif
    }
#if ASYNC_LDS_OK
#if __has_builtin(__builtin_amdgcn_s_wait_asynccnt)
    __builtin_amdgcn_s_wait_asynccnt(0);
#else
    asm volatile("s_wait_asynccnt 0x0" ::: "memory");
#endif
#endif
    __syncthreads();

    const int wave = tid >> 5, lane = tid & 31;
    const int lm = lane & 15, lh = lane >> 4;

    const unsigned short* up_e = upT + (size_t)e * INTER_ * HD_;  // [INTER][HD]
    const unsigned short* dn_e = dnT + (size_t)e * HD_ * INTER_;  // [HD][INTER]
    const unsigned short* ha = &s_h[lm * SH_LDW];                 // this lane's A row

    f32x8 acc[4] = {};
    for (int ic = 0; ic < INTER_; ic += 128) {
        // L2 prefetch for this chunk's down-proj weight rows
        for (int nt = 0; nt < 4; ++nt)
            __builtin_prefetch(dn_e + (size_t)(wave * 64 + nt * 16 + lm) * INTER_ + ic, 0, 3);

        // ---- up-proj: this wave computes act columns [ic+16*wave, +16), pipelined
        const unsigned short* brow = up_e + (size_t)(ic + wave * 16 + lm) * HD_;
        f32x8 ua = {};
        u32x8 ub_cur = load_b_raw(brow, 0, lh);
        for (int k0 = 0; k0 < HD_ - 32; k0 += 32) {
            u32x8 ub_nxt = load_b_raw(brow, k0 + 32, lh);
            ua = wmma_bf16(load_a_raw(ha, k0, lh), ub_cur, ua);
            ub_cur = ub_nxt;
        }
        ua = wmma_bf16(load_a_raw(ha, HD_ - 32, lh), ub_cur, ua);

        __syncthreads();  // protect previous chunk's s_act readers
        {
            int il = wave * 16 + lm;
            for (int v = 0; v < 8; ++v) {
                int m = v + 8 * lh;
                float r = fmaxf(ua[v], 0.0f);
                s_act[m * ACT_LDW + il] = f2bf(r * r);
            }
        }
        __syncthreads();

        // ---- down-proj: 4 n-subtiles, K=128 local, pipelined B stream
        const unsigned short* drow[4];
        for (int nt = 0; nt < 4; ++nt)
            drow[nt] = dn_e + (size_t)(wave * 64 + nt * 16 + lm) * INTER_ + ic;
        u32x8 db_cur[4];
        for (int nt = 0; nt < 4; ++nt) db_cur[nt] = load_b_raw(drow[nt], 0, lh);
        for (int kk = 0; kk < 96; kk += 32) {
            u32x8 db_nxt[4];
            for (int nt = 0; nt < 4; ++nt) db_nxt[nt] = load_b_raw(drow[nt], kk + 32, lh);
            u32x8 af = load_a_raw(&s_act[lm * ACT_LDW], kk, lh);
            for (int nt = 0; nt < 4; ++nt) acc[nt] = wmma_bf16(af, db_cur[nt], acc[nt]);
            for (int nt = 0; nt < 4; ++nt) db_cur[nt] = db_nxt[nt];
        }
        u32x8 af = load_a_raw(&s_act[lm * ACT_LDW], 96, lh);
        for (int nt = 0; nt < 4; ++nt) acc[nt] = wmma_bf16(af, db_cur[nt], acc[nt]);
    }

    // weighted scatter into per-(token, slot) buffer
    for (int nt = 0; nt < 4; ++nt) {
        int n = wave * 64 + nt * 16 + lm;
        for (int v = 0; v < 8; ++v) {
            int m = v + 8 * lh;
            if (s_slot[m] >= 0) {
                size_t o = (((size_t)s_tok[m] * 2 + s_slot[m]) << 9) + n;
                slots[o] = s_w[m] * acc[nt][v];
            }
        }
    }
}

// ---------------- combine the two expert slots per token ----------------
__global__ void k_combine(const float* __restrict__ slots,
                          unsigned short* __restrict__ comb, long n) {
    for (long o = (long)blockIdx.x * blockDim.x + threadIdx.x; o < n;
         o += (long)gridDim.x * blockDim.x) {
        long t = o >> 9;
        int  c = (int)(o & 511);
        float v = slots[(t << 10) + c] + slots[(t << 10) + 512 + c];
        comb[o] = f2bf(v);
    }
}

// ---------------- host-side orchestration ----------------
extern "C" void kernel_launch(void* const* d_in, const int* in_sizes, int n_in,
                              void* d_out, int out_size, void* d_ws, size_t ws_size,
                              hipStream_t stream) {
    (void)in_sizes; (void)n_in; (void)out_size; (void)ws_size;
    const float* x      = (const float*)d_in[0];
    const float* head_w = (const float*)d_in[1];
    const float* head_b = (const float*)d_in[2];
    const float* out_w  = (const float*)d_in[3];
    const float* out_b  = (const float*)d_in[4];
    const float* emb    = (const float*)d_in[5];
    const float* up_w   = (const float*)d_in[6];
    const float* down_w = (const float*)d_in[7];
    float* out = (float*)d_out;

    char* ws = (char*)d_ws;
    unsigned short* xbf    = (unsigned short*)(ws + 0);           // 16 MB
    unsigned short* hwbf   = (unsigned short*)(ws + 16777216);    //  8 MB
    unsigned short* owbf   = (unsigned short*)(ws + 25165824);    //  8 MB
    unsigned short* upT    = (unsigned short*)(ws + 33554432);    // 16 MB [E,INTER,HD]
    unsigned short* dnT    = (unsigned short*)(ws + 50331648);    // 16 MB [E,HD,INTER]
    unsigned short* hbf    = (unsigned short*)(ws + 67108864);    // 16 MB [TOKENS,HD]
    unsigned short* combbf = (unsigned short*)(ws + 83886080);    // 16 MB
    float*          slots  = (float*)(ws + 100663296);            // 64 MB [TOKENS,2,HD]
    int*            cnt    = (int*)(ws + 167772160);
    int*            tokl   = (int*)(ws + 167772416);              // 512 KB
    float*          wl     = (float*)(ws + 168296704);            // 512 KB
    int*            sl     = (int*)(ws + 168820992);              // 512 KB

    // 1) precision conversion (+ weight transposes so WMMA B-operands are contiguous)
    k_convert_bf16<<<4096, 256, 0, stream>>>(x, xbf, (long)BS_ * H_);
    k_convert_bf16<<<4096, 256, 0, stream>>>(head_w, hwbf, (long)H_ * H_);
    k_convert_bf16<<<4096, 256, 0, stream>>>(out_w, owbf, (long)H_ * H_);
    k_convert_transpose_bf16<<<4096, 256, 0, stream>>>(up_w, upT, HD_, INTER_,
                                                       (long)NEXP_ * HD_ * INTER_);
    k_convert_transpose_bf16<<<4096, 256, 0, stream>>>(down_w, dnT, INTER_, HD_,
                                                       (long)NEXP_ * INTER_ * HD_);
    k_zero_cnt<<<1, 32, 0, stream>>>(cnt);

    // 2) head projection: h = x @ head_w^T + head_b  (bf16 out, viewed [TOKENS,HD])
    k_gemm_nt<1><<<dim3(BS_ / 64, H_ / 128), 256, 0, stream>>>(xbf, hwbf, head_b, hbf,
                                                               BS_, H_, H_);
    // 3) router + top-2 compaction
    k_router<<<TOKENS_ / 4, 128, 0, stream>>>(hbf, emb, cnt, tokl, wl, sl);

    // 4) fused expert FFN, only selected (token, expert) pairs
    k_ffn_expert<<<dim3(TOKENS_ / 16, NEXP_), 256, 0, stream>>>(hbf, upT, dnT, cnt,
                                                                tokl, wl, sl, slots);
    // 5) combine slots -> comb (bf16)
    k_combine<<<8192, 256, 0, stream>>>(slots, combbf, (long)TOKENS_ * HD_);

    // 6) output projection: y = comb @ out_w^T + out_b (f32 out)
    k_gemm_nt<0><<<dim3(BS_ / 64, H_ / 128), 256, 0, stream>>>(combbf, owbf, out_b, out,
                                                               BS_, H_, H_);
}
